// SwitchTopKSelector_24824910970895
// MI455X (gfx1250) — matrix operand order repaired and checked by reading, not compiled, not run on hardware
//
#include <hip/hip_runtime.h>
#include <hip/hip_bf16.h>

#define N_TOKENS   16384
#define D_MODEL    2048
#define N_EXPERTS  64
#define KCHUNK     32
#define TOK_PER_BLOCK 64
#define THREADS    128
#define WAVES      4
#define LPITCH     68   // floats per token row: 16B-aligned rows, conflict-free banks

typedef __attribute__((ext_vector_type(16))) __bf16 v16bf;
typedef __attribute__((ext_vector_type(8)))  float  v8f;

union BF16Frag {
  v16bf v;
  uint4 q[2];
};

__device__ __forceinline__ float softplus_f(float x) {
  // branchless, numerically stable: max(x,0) + log(1 + e^-|x|)
  return fmaxf(x, 0.f) + __logf(1.f + __expf(-fabsf(x)));
}

// ---------------------------------------------------------------------------
// Prep: convert W_gate / W_noise (f32, [K=2048][E=64]) once into d_ws as bf16
// in expert-major [E][K] order == exactly the WMMA B-fragment streaming layout.
// Also zeroes the per-expert probability accumulator.
// ---------------------------------------------------------------------------
__global__ __launch_bounds__(256)
void prep_kernel(const float* __restrict__ Wg,
                 const float* __restrict__ Wn,
                 __bf16* __restrict__ gbf,
                 __bf16* __restrict__ nbf,
                 float* __restrict__ es)
{
  int o = blockIdx.x * 256 + threadIdx.x;       // 0 .. 64*2048-1
  int k = o & (D_MODEL - 1);
  int e = o >> 11;
  gbf[o] = (__bf16)Wg[(size_t)k * N_EXPERTS + e];
  nbf[o] = (__bf16)Wn[(size_t)k * N_EXPERTS + e];
  if (o < N_EXPERTS) es[o] = 0.f;
}

// ---------------------------------------------------------------------------
// Main fused router kernel: dual bf16 WMMA GEMM (gate + noise projections)
// + noisy-logit epilogue + top-2 masked softmax + per-expert prob reduction.
// Block = 128 threads (4 waves); each wave owns a 16-token M-tile, full K,
// all 4 expert N-tiles for both projections. No barriers inside the k-loop.
// ---------------------------------------------------------------------------
__global__ __launch_bounds__(THREADS)
void moe_router_kernel(const float* __restrict__ x,
                       const __bf16* __restrict__ wgbf,   // [64][2048] bf16
                       const float* __restrict__ bg,
                       const __bf16* __restrict__ wnbf,   // [64][2048] bf16
                       const float* __restrict__ bn,
                       const float* __restrict__ noise,
                       float* __restrict__ out,
                       float* __restrict__ expert_sums)
{
  __shared__ __align__(16) float sLog[TOK_PER_BLOCK * LPITCH];

  const int tid  = threadIdx.x;
  const int lane = tid & 31;
  const int wave = tid >> 5;
  const int hf   = lane >> 4;   // wave half
  const int l16  = lane & 15;
  const int blockTok = blockIdx.x * TOK_PER_BLOCK;
  const int waveTok  = blockTok + wave * 16;

  // pull this block's noise tile (16 KB) toward the caches early
  __builtin_prefetch(noise + (size_t)blockTok * N_EXPERTS + tid * 32, 0, 0);

  const v8f vzero = {0.f,0.f,0.f,0.f,0.f,0.f,0.f,0.f};
  v8f accG[4], accN[4];
#pragma unroll
  for (int i = 0; i < 4; ++i) { accG[i] = vzero; accN[i] = vzero; }

  // A row: token = waveTok + l16; K-halves per the 16-bit A-fragment layout
  const float* aRow = x + (size_t)(waveTok + l16) * D_MODEL + hf * 8;
  // B rows: lane owns expert column e = nt*16 + l16, K-half selected by hf
  const __bf16* bgRow = wgbf + (size_t)l16 * D_MODEL + hf * 8;
  const __bf16* bnRow = wnbf + (size_t)l16 * D_MODEL + hf * 8;

  for (int kc = 0; kc < D_MODEL; kc += KCHUNK) {
    // ---- A fragment: lanes 0-15 hold K {0..7, 16..23}, lanes 16-31 {8..15, 24..31}
    BF16Frag a;
    {
      const float* ap = aRow + kc;
      float4 f0 = ((const float4*)ap)[0];
      float4 f1 = ((const float4*)(ap + 4))[0];
      float4 f2 = ((const float4*)(ap + 16))[0];
      float4 f3 = ((const float4*)(ap + 20))[0];
      a.v[0]  = (__bf16)f0.x; a.v[1]  = (__bf16)f0.y;
      a.v[2]  = (__bf16)f0.z; a.v[3]  = (__bf16)f0.w;
      a.v[4]  = (__bf16)f1.x; a.v[5]  = (__bf16)f1.y;
      a.v[6]  = (__bf16)f1.z; a.v[7]  = (__bf16)f1.w;
      a.v[8]  = (__bf16)f2.x; a.v[9]  = (__bf16)f2.y;
      a.v[10] = (__bf16)f2.z; a.v[11] = (__bf16)f2.w;
      a.v[12] = (__bf16)f3.x; a.v[13] = (__bf16)f3.y;
      a.v[14] = (__bf16)f3.z; a.v[15] = (__bf16)f3.w;
    }

    // ---- 4 N-tiles x {gate, noise}: B fragments stream straight from the
    // ---- L2-resident bf16 tables (two 16-byte loads per lane per fragment, at byte offsets 0 and 32)
#pragma unroll
    for (int nt = 0; nt < 4; ++nt) {
      BF16Frag b;
      const __bf16* bp = bgRow + (size_t)(nt * 16) * D_MODEL + kc;
      b.q[0] = ((const uint4*)bp)[0];
      b.q[1] = ((const uint4*)bp)[2];
      accG[nt] = __builtin_amdgcn_wmma_f32_16x16x32_bf16(
          false, a.v, false, b.v, (short)0, accG[nt], false, false);

      const __bf16* bp2 = bnRow + (size_t)(nt * 16) * D_MODEL + kc;
      b.q[0] = ((const uint4*)bp2)[0];
      b.q[1] = ((const uint4*)bp2)[2];
      accN[nt] = __builtin_amdgcn_wmma_f32_16x16x32_bf16(
          false, a.v, false, b.v, (short)0, accN[nt], false, false);
    }
  }

  // ---- combine into noisy logits (C-layout: row m = v + 8*hf, col e = nt*16+l16)
  float bgv[4], bnv[4];
#pragma unroll
  for (int nt = 0; nt < 4; ++nt) {
    bgv[nt] = bg[nt * 16 + l16];
    bnv[nt] = bn[nt * 16 + l16];
  }
#pragma unroll
  for (int nt = 0; nt < 4; ++nt) {
#pragma unroll
    for (int v = 0; v < 8; ++v) {
      int m = v + hf * 8;
      int t = waveTok + m;
      int e = nt * 16 + l16;
      float nz = noise[(size_t)t * N_EXPERTS + e];
      float logit = accG[nt][v] + bgv[nt] + nz * softplus_f(accN[nt][v] + bnv[nt]);
      sLog[(wave * 16 + m) * LPITCH + e] = logit;
    }
  }
  __syncthreads();

  // ---- per-token epilogue: lanes 0-15 of each wave own one token each.
  // Row pointers are 16B-aligned (LPITCH*4 = 272 = 16*17) -> b128 LDS traffic.
  if (lane < 16) {
    int t = waveTok + lane;
    float* row = &sLog[(wave * 16 + lane) * LPITCH];
    float4* rowq = (float4*)row;

    // pass 1: top-2 scan
    float m1 = -3.402823466e38f, m2 = -3.402823466e38f;
    int i1 = 0, i2 = 0;
#pragma unroll 4
    for (int j = 0; j < N_EXPERTS / 4; ++j) {
      float4 q = rowq[j];
      float vv[4] = {q.x, q.y, q.z, q.w};
#pragma unroll
      for (int c = 0; c < 4; ++c) {
        float val = vv[c];
        int e = j * 4 + c;
        if (val > m1) { m2 = m1; i2 = i1; m1 = val; i1 = e; }
        else if (val > m2) { m2 = val; i2 = e; }
      }
    }

    // pass 2: exp(v - m1), row <- exp values
    float s = 0.f;
#pragma unroll 4
    for (int j = 0; j < N_EXPERTS / 4; ++j) {
      float4 q = rowq[j];
      q.x = __expf(q.x - m1); q.y = __expf(q.y - m1);
      q.z = __expf(q.z - m1); q.w = __expf(q.w - m1);
      s += q.x + q.y + q.z + q.w;
      rowq[j] = q;
    }

    // pass 3: normalize in place -> row holds softmax probs for aux reduction
    float invs = 1.f / s;
#pragma unroll 4
    for (int j = 0; j < N_EXPERTS / 4; ++j) {
      float4 q = rowq[j];
      q.x *= invs; q.y *= invs; q.z *= invs; q.w *= invs;
      rowq[j] = q;
    }

    // top-2 masked softmax -> selection weights (row-max is always top-1)
    float bexp = __expf(m2 - m1);
    float denom = 1.f + bexp;
    float w1 = 1.f / denom;
    float w2 = bexp / denom;

    float* orow = out + (size_t)t * N_EXPERTS;
#pragma unroll
    for (int j = 0; j < N_EXPERTS / 4; ++j) {
      int e0 = j * 4;
      float4 z;
      z.x = (e0     == i1) ? w1 : ((e0     == i2) ? w2 : 0.f);
      z.y = (e0 + 1 == i1) ? w1 : ((e0 + 1 == i2) ? w2 : 0.f);
      z.z = (e0 + 2 == i1) ? w1 : ((e0 + 2 == i2) ? w2 : 0.f);
      z.w = (e0 + 3 == i1) ? w1 : ((e0 + 3 == i2) ? w2 : 0.f);
      ((float4*)orow)[j] = z;
    }
  }
  __syncthreads();

  // ---- per-expert prob sums: conflict-free column sum over the 64-token tile,
  // one global atomic per expert per block (no LDS atomics).
  if (tid < N_EXPERTS) {
    float s = 0.f;
#pragma unroll 4
    for (int t = 0; t < TOK_PER_BLOCK; ++t)
      s += sLog[t * LPITCH + tid];
    atomicAdd(&expert_sums[tid], s);
  }
}

__global__ void aux_loss_kernel(const float* __restrict__ es, float* __restrict__ out) {
  __shared__ float red[N_EXPERTS];
  int tid = threadIdx.x;
  float m = es[tid] * (1.f / (float)N_TOKENS);
  red[tid] = m * m;
  __syncthreads();
  for (int off = N_EXPERTS / 2; off > 0; off >>= 1) {
    if (tid < off) red[tid] += red[tid + off];
    __syncthreads();
  }
  if (tid == 0) out[(size_t)N_TOKENS * N_EXPERTS] = red[0] * (float)N_EXPERTS;
}

extern "C" void kernel_launch(void* const* d_in, const int* in_sizes, int n_in,
                              void* d_out, int out_size, void* d_ws, size_t ws_size,
                              hipStream_t stream) {
  const float* x     = (const float*)d_in[0];
  const float* Wg    = (const float*)d_in[1];
  const float* bg    = (const float*)d_in[2];
  const float* Wn    = (const float*)d_in[3];
  const float* bn    = (const float*)d_in[4];
  const float* noise = (const float*)d_in[5];
  float* out = (float*)d_out;

  // workspace layout: [0,256) es; [256, 256+256K) Wg bf16; then Wn bf16
  char* ws = (char*)d_ws;
  float*  es   = (float*)ws;
  __bf16* gbf  = (__bf16*)(ws + 256);
  __bf16* nbf  = (__bf16*)(ws + 256 + (size_t)N_EXPERTS * D_MODEL * sizeof(__bf16));

  prep_kernel<<<(N_EXPERTS * D_MODEL) / 256, 256, 0, stream>>>(Wg, Wn, gbf, nbf, es);
  moe_router_kernel<<<N_TOKENS / TOK_PER_BLOCK, THREADS, 0, stream>>>(
      x, gbf, bg, nbf, bn, noise, out, es);
  aux_loss_kernel<<<1, N_EXPERTS, 0, stream>>>(es, out);
}
